// GCNConvModel_21981642620995
// MI455X (gfx1250) — matrix operand order repaired and checked by reading, not compiled, not run on hardware
//
#include <hip/hip_runtime.h>
#include <hip/hip_bf16.h>

#define IN_CH_   1024
#define FEAT_CH_ 128
#define OUT_CH_  3

typedef __attribute__((ext_vector_type(4)))  float  v4f;
typedef __attribute__((ext_vector_type(8)))  unsigned short v8u;
typedef __attribute__((ext_vector_type(8)))  __bf16 v8bf;
typedef __attribute__((ext_vector_type(16))) __bf16 v16bf;
typedef __attribute__((ext_vector_type(8)))  float  v8f;

__device__ __forceinline__ unsigned short f2bf_rne(float f) {
    unsigned int u = __float_as_uint(f);
    u += 0x7FFFu + ((u >> 16) & 1u);   // round-to-nearest-even to bf16
    return (unsigned short)(u >> 16);
}

// ---------------------------------------------------------------------------
// Pack W1 (1024x128 f32) into bf16 in B-fragment register order:
//   Bp[(((strip*32 + t)*32 + lane)*16 + e)] = bf16(W1[k][c])
//   k = 32*t + 8*(lane>>4) + (e&7) + 16*(e>>3) ; c = strip*16 + (lane&15)
// so a wave's B fragment for (strip, K-step t) is one 32B contiguous chunk/lane.
// ---------------------------------------------------------------------------
__global__ void k_pack_w1(const float* __restrict__ W, unsigned short* __restrict__ Bp) {
    int o = blockIdx.x * blockDim.x + threadIdx.x;
    if (o >= IN_CH_ * FEAT_CH_) return;
    int e     = o & 15;
    int lane  = (o >> 4) & 31;
    int t     = (o >> 9) & 31;
    int strip = o >> 14;
    int k = 32 * t + 8 * (lane >> 4) + (e & 7) + 16 * (e >> 3);
    int c = strip * 16 + (lane & 15);
    Bp[o] = f2bf_rne(W[(size_t)k * FEAT_CH_ + c]);
}

// ---------------------------------------------------------------------------
// GEMM1: H[N,128] = bf16(X[N,1024]) @ bf16(W1) via v_wmma_f32_16x16x32_bf16.
// Block = 8 waves, each wave owns its own 16-row M-subtile (block = 128 rows)
// and all 8 N-strips (8 accumulators). Per K-step the sched_group_barriers
// pin the pipeline: all VMEM reads (A + 8 B fragments) issue as one batch,
// then the 8-WMMA chain runs behind a single wait. No LDS, no barriers.
// ---------------------------------------------------------------------------
__global__ __launch_bounds__(256) void gemm1_wmma(
    const float* __restrict__ X, const unsigned short* __restrict__ Bp,
    float* __restrict__ H, int n) {
    const int lane = threadIdx.x;                 // 0..31
    const int wave = threadIdx.y;                 // 0..7
    const int m0   = blockIdx.x * 128 + wave * 16;
    if (m0 >= n) return;                          // wave-uniform exit (EXEC stays all-1)
    const int half = lane >> 4;
    const int l16  = lane & 15;

    v8f acc[8] = {};
    // Lane's A row; A-frag layout: e<8 -> K=k0+8*half+e ; e>=8 -> K=k0+16+8*half+(e-8)
    const float* rowp = X + (size_t)(m0 + l16) * IN_CH_ + 8 * half;
    // Lane's B base: fragments at stride 32*32*16 ushorts per strip
    const unsigned short* bbase = Bp + (size_t)lane * 16;

    for (int t = 0; t < 32; ++t) {                // k0 = 32*t
        const float* rp = rowp + 32 * t;

        // --- A fragment: two contiguous 32B chunks of the lane's row ---
        v4f a0 = ((const v4f*)rp)[0];
        v4f a1 = ((const v4f*)rp)[1];
        v4f a2 = ((const v4f*)(rp + 16))[0];
        v4f a3 = ((const v4f*)(rp + 16))[1];

        // --- All 16 B fragment loads (kept live; issued as one batch) ---
        v8bf blo[8], bhi[8];
        #pragma unroll
        for (int n0 = 0; n0 < 8; ++n0) {
            const v8u* bp = (const v8u*)(bbase + ((size_t)n0 * 32 + t) * 512);
            blo[n0] = __builtin_bit_cast(v8bf, bp[0]);
            bhi[n0] = __builtin_bit_cast(v8bf, bp[1]);
        }

        // --- Convert A to bf16 (overlaps with B loads in flight) ---
        union { v16bf v; unsigned short u[16]; } af;
        #pragma unroll
        for (int e = 0; e < 4; ++e) {
            af.u[e]      = f2bf_rne(a0[e]);
            af.u[4 + e]  = f2bf_rne(a1[e]);
            af.u[8 + e]  = f2bf_rne(a2[e]);
            af.u[12 + e] = f2bf_rne(a3[e]);
        }

        // --- 8 WMMAs back-to-back, all sharing one A fragment ---
        #pragma unroll
        for (int n0 = 0; n0 < 8; ++n0) {
            v16bf b = __builtin_shufflevector(blo[n0], bhi[n0],
                0,1,2,3,4,5,6,7,8,9,10,11,12,13,14,15);
            acc[n0] = __builtin_amdgcn_wmma_f32_16x16x32_bf16(
                false, af.v, false, b, (short)0, acc[n0], false, false);
        }

#if __has_builtin(__builtin_amdgcn_sched_group_barrier)
        // Pin per-iteration pipeline: VMEM reads first, then the WMMA chain.
        __builtin_amdgcn_sched_group_barrier(0x0020, 20, 0); // VMEM reads
        __builtin_amdgcn_sched_group_barrier(0x0002, 64, 0); // VALU (bf16 converts)
        __builtin_amdgcn_sched_group_barrier(0x0008, 8, 0);  // WMMA chain
#endif
    }

    // C/D layout: VGPR r holds row m0 + r + 8*half, col = n0*16 + l16
    #pragma unroll
    for (int n0 = 0; n0 < 8; ++n0) {
        #pragma unroll
        for (int r = 0; r < 8; ++r)
            H[(size_t)(m0 + r + 8 * half) * FEAT_CH_ + n0 * 16 + l16] = acc[n0][r];
    }
}

// ---------------------------------------------------------------------------
// Graph-normalization helpers
// ---------------------------------------------------------------------------
__global__ void k_init_deg(float* deg, int n) {
    int i = blockIdx.x * blockDim.x + threadIdx.x;
    if (i < n) deg[i] = 1.0f;                     // self-loop
}

__global__ void k_count_deg(const int* __restrict__ dst, float* deg, int e) {
    int i = blockIdx.x * blockDim.x + threadIdx.x;
    if (i < e) atomicAdd(&deg[dst[i]], 1.0f);
}

__global__ void k_rsqrt_inplace(float* d, int n) {
    int i = blockIdx.x * blockDim.x + threadIdx.x;
    if (i < n) d[i] = rsqrtf(d[i]);
}

__global__ void k_zero(float* p, long long n) {
    long long i = (long long)blockIdx.x * blockDim.x + threadIdx.x;
    if (i < n) p[i] = 0.0f;
}

// Edge scatter layer 1: Y1[dst] += H[src] * dinv[src]*dinv[dst]
// 32 threads per edge, 4 features each; H is L2-resident (51 MB < 192 MB).
__global__ void k_scatter1(const int* __restrict__ src, const int* __restrict__ dst,
                           const float* __restrict__ dinv, const float* __restrict__ H,
                           float* __restrict__ Y1, int E) {
    long long t = (long long)blockIdx.x * blockDim.x + threadIdx.x;
    if (t >= (long long)E * 32) return;
    int e = (int)(t >> 5), g = (int)(t & 31);
    int s = src[e], d = dst[e];
    float norm = dinv[s] * dinv[d];
    const float* hr = H  + (size_t)s * FEAT_CH_ + g * 4;
    float*       yr = Y1 + (size_t)d * FEAT_CH_ + g * 4;
    atomicAdd(yr + 0, hr[0] * norm);
    atomicAdd(yr + 1, hr[1] * norm);
    atomicAdd(yr + 2, hr[2] * norm);
    atomicAdd(yr + 3, hr[3] * norm);
}

// Add self-loop term + bias, relu (in place Y1 -> X1)
__global__ void k_finalize1(float* __restrict__ Y1, const float* __restrict__ H,
                            const float* __restrict__ dinv, const float* __restrict__ b1,
                            int n) {
    long long t = (long long)blockIdx.x * blockDim.x + threadIdx.x;
    if (t >= (long long)n * FEAT_CH_) return;
    int i = (int)(t >> 7), f = (int)(t & 127);
    float s = dinv[i];
    float v = Y1[t] + H[t] * s * s + b1[f];
    Y1[t] = v > 0.0f ? v : 0.0f;
}

// GEMM2: H2[N,3] = X1[N,128] @ W2[128,3]; one wave per node, shuffle-reduce.
__global__ __launch_bounds__(256) void k_gemm2(const float* __restrict__ X1,
                                               const float* __restrict__ W2,
                                               float* __restrict__ H2, int n) {
    int gw   = (blockIdx.x * blockDim.x + threadIdx.x) >> 5;  // wave id = node
    int lane = threadIdx.x & 31;
    if (gw >= n) return;                                      // uniform per wave
    const float* row = X1 + (size_t)gw * FEAT_CH_;
    float a0 = 0.f, a1 = 0.f, a2 = 0.f;
    #pragma unroll
    for (int j = 0; j < 4; ++j) {
        int k = lane + j * 32;
        float x = row[k];
        a0 += x * W2[k * 3 + 0];
        a1 += x * W2[k * 3 + 1];
        a2 += x * W2[k * 3 + 2];
    }
    #pragma unroll
    for (int off = 16; off > 0; off >>= 1) {
        a0 += __shfl_down(a0, off, 32);
        a1 += __shfl_down(a1, off, 32);
        a2 += __shfl_down(a2, off, 32);
    }
    if (lane == 0) {
        H2[(size_t)gw * 3 + 0] = a0;
        H2[(size_t)gw * 3 + 1] = a1;
        H2[(size_t)gw * 3 + 2] = a2;
    }
}

// out = self-loop term + bias (initializes every element of d_out)
__global__ void k_init_out(float* __restrict__ out, const float* __restrict__ H2,
                           const float* __restrict__ dinv, const float* __restrict__ b2,
                           int n) {
    int t = blockIdx.x * blockDim.x + threadIdx.x;
    if (t >= n * OUT_CH_) return;
    int i = t / OUT_CH_, c = t - i * OUT_CH_;
    float s = dinv[i];
    out[t] = H2[t] * s * s + b2[c];
}

__global__ void k_scatter2(const int* __restrict__ src, const int* __restrict__ dst,
                           const float* __restrict__ dinv, const float* __restrict__ H2,
                           float* __restrict__ out, int E) {
    int e = blockIdx.x * blockDim.x + threadIdx.x;
    if (e >= E) return;
    int s = src[e], d = dst[e];
    float norm = dinv[s] * dinv[d];
    atomicAdd(&out[(size_t)d * 3 + 0], H2[(size_t)s * 3 + 0] * norm);
    atomicAdd(&out[(size_t)d * 3 + 1], H2[(size_t)s * 3 + 1] * norm);
    atomicAdd(&out[(size_t)d * 3 + 2], H2[(size_t)s * 3 + 2] * norm);
}

extern "C" void kernel_launch(void* const* d_in, const int* in_sizes, int n_in,
                              void* d_out, int out_size, void* d_ws, size_t ws_size,
                              hipStream_t stream) {
    const float* features = (const float*)d_in[0];
    const int*   edges2   = (const int*)d_in[2];
    const float* W1       = (const float*)d_in[5];
    const float* b1       = (const float*)d_in[6];
    const float* W2       = (const float*)d_in[7];
    const float* b2       = (const float*)d_in[8];
    float* out = (float*)d_out;

    const int N = in_sizes[0] / IN_CH_;   // 100000 (multiple of 16)
    const int E = in_sizes[2] / 2;        // 1600000
    const int* src = edges2;              // edges2[0,:]
    const int* dst = edges2 + E;          // edges2[1,:]

    // Workspace layout (float units, 256-aligned sections)
    float* dinv = (float*)d_ws;                                   // N
    float* H    = dinv + (((size_t)N + 255) & ~(size_t)255);      // N*128
    float* Y1   = H + (size_t)N * FEAT_CH_;                       // N*128 (becomes X1)
    float* H2   = Y1 + (size_t)N * FEAT_CH_;                      // N*3
    unsigned short* Bp = (unsigned short*)                        // 1024*128 bf16
        (H2 + ((((size_t)N * OUT_CH_) + 255) & ~(size_t)255));

    const int T = 256;
    long long nf = (long long)N * FEAT_CH_;

    // Degree / normalization (shared by both layers)
    k_init_deg<<<(N + T - 1) / T, T, 0, stream>>>(dinv, N);
    k_count_deg<<<(E + T - 1) / T, T, 0, stream>>>(dst, dinv, E);
    k_rsqrt_inplace<<<(N + T - 1) / T, T, 0, stream>>>(dinv, N);

    // Layer 1
    k_pack_w1<<<(IN_CH_ * FEAT_CH_ + T - 1) / T, T, 0, stream>>>(W1, Bp);
    k_zero<<<(unsigned)((nf + T - 1) / T), T, 0, stream>>>(Y1, nf);
    gemm1_wmma<<<(N + 127) / 128, dim3(32, 8), 0, stream>>>(features, Bp, H, N);
    k_scatter1<<<(unsigned)(((long long)E * 32 + T - 1) / T), T, 0, stream>>>(
        src, dst, dinv, H, Y1, E);
    k_finalize1<<<(unsigned)((nf + T - 1) / T), T, 0, stream>>>(Y1, H, dinv, b1, N);

    // Layer 2
    k_gemm2<<<(unsigned)(((long long)N * 32 + T - 1) / T), T, 0, stream>>>(Y1, W2, H2, N);
    k_init_out<<<(N * OUT_CH_ + T - 1) / T, T, 0, stream>>>(out, H2, dinv, b2, N);
    k_scatter2<<<(E + T - 1) / T, T, 0, stream>>>(src, dst, dinv, H2, out, E);
}
